// AttentionBlock_88965952569373
// MI455X (gfx1250) — compile-verified, matrix operand order checked by reading, not run on hardware
//
#include <hip/hip_runtime.h>

typedef __attribute__((ext_vector_type(16))) __bf16 v16bf;
typedef __attribute__((ext_vector_type(8)))  __bf16 v8bf;
typedef __attribute__((ext_vector_type(8)))  float  v8f;

static constexpr int Bsz  = 8;
static constexpr int Cdim = 512;
static constexpr int Tdim = 2048;
static constexpr int KD   = 512;   // KEY_SIZE
static constexpr int VD   = 512;   // VALUE_SIZE

// ---------- bf16 helpers ----------
__device__ __forceinline__ unsigned short f32_bf16(float f) {
  union { float f; unsigned u; } v; v.f = f;
  unsigned u = v.u;
  u += 0x7FFFu + ((u >> 16) & 1u);   // round-to-nearest-even
  return (unsigned short)(u >> 16);
}
__device__ __forceinline__ __bf16 bits_bf16(unsigned short h) {
  union { unsigned short u; __bf16 b; } v; v.u = h; return v.b;
}

// ---------- WMMA fragment loaders ----------
// 16-bit A (16x32) / B-as-rows (32x16) layout: lane L holds row (L&15),
// K halves at (L>>4)*8 and (L>>4)*8+16 (8 contiguous bf16 = 16B each).
__device__ __forceinline__ v16bf load_frag_bf16(const unsigned short* p0, int ld, int lane) {
  const unsigned short* p = p0 + (size_t)(lane & 15) * ld + ((lane >> 4) << 3);
  v8bf lo = *(const v8bf*)(p);
  v8bf hi = *(const v8bf*)(p + 16);
  v16bf r;
#pragma unroll
  for (int i = 0; i < 8; i++) { r[i] = lo[i]; r[i + 8] = hi[i]; }
  return r;
}
// Same layout, source is fp32 row-major, convert in-register.
__device__ __forceinline__ v16bf load_frag_f32(const float* p0, int ld, int lane) {
  const float* p = p0 + (size_t)(lane & 15) * ld + ((lane >> 4) << 3);
  v16bf r;
#pragma unroll
  for (int i = 0; i < 8; i++) {
    r[i]     = bits_bf16(f32_bf16(p[i]));
    r[i + 8] = bits_bf16(f32_bf16(p[i + 16]));
  }
  return r;
}

// ---------- wave-level 32x64 GEMM cores (2x4 tiles of 16x16) ----------
template <int MT, int NT>
__device__ __forceinline__ void gemm_bf16(const unsigned short* __restrict__ A, int lda,
                                          const unsigned short* __restrict__ Bp, int ldb,
                                          int K, int lane, v8f (&acc)[MT][NT]) {
  for (int k = 0; k < K; k += 32) {
    v16bf af[MT], bfv[NT];
#pragma unroll
    for (int i = 0; i < MT; i++) af[i] = load_frag_bf16(A + (size_t)(i * 16) * lda + k, lda, lane);
#pragma unroll
    for (int j = 0; j < NT; j++) bfv[j] = load_frag_bf16(Bp + (size_t)(j * 16) * ldb + k, ldb, lane);
    if (k + 32 < K) {
      __builtin_prefetch(A + k + 32, 0, 1);
      __builtin_prefetch(Bp + k + 32, 0, 1);
    }
#pragma unroll
    for (int i = 0; i < MT; i++)
#pragma unroll
      for (int j = 0; j < NT; j++)
        acc[i][j] = __builtin_amdgcn_wmma_f32_16x16x32_bf16(
            false, af[i], false, bfv[j], (short)0, acc[i][j], false, false);
  }
}

template <int MT, int NT>
__device__ __forceinline__ void gemm_f32A(const float* __restrict__ A, int lda,
                                          const unsigned short* __restrict__ Bp, int ldb,
                                          int K, int lane, v8f (&acc)[MT][NT]) {
  for (int k = 0; k < K; k += 32) {
    v16bf af[MT], bfv[NT];
#pragma unroll
    for (int i = 0; i < MT; i++) af[i] = load_frag_f32(A + (size_t)(i * 16) * lda + k, lda, lane);
#pragma unroll
    for (int j = 0; j < NT; j++) bfv[j] = load_frag_bf16(Bp + (size_t)(j * 16) * ldb + k, ldb, lane);
#pragma unroll
    for (int i = 0; i < MT; i++)
#pragma unroll
      for (int j = 0; j < NT; j++)
        acc[i][j] = __builtin_amdgcn_wmma_f32_16x16x32_bf16(
            false, af[i], false, bfv[j], (short)0, acc[i][j], false, false);
  }
}

// ---------- kernel 0: LDS-tiled transpose + f32->bf16 convert ----------
// in [rows, cols] f32  ->  out [cols, rows] bf16   (block 32x8, tile 32x32)
__global__ __launch_bounds__(256) void k_transpose_cvt(const float* __restrict__ in,
                                                       unsigned short* __restrict__ out,
                                                       int rows, int cols,
                                                       long long in_bs, long long out_bs) {
  __shared__ unsigned short tile[32][33];
  const float* ip = in + (size_t)blockIdx.z * in_bs;
  unsigned short* op = out + (size_t)blockIdx.z * out_bs;
  int c0 = blockIdx.x * 32, r0 = blockIdx.y * 32;
  int tx = threadIdx.x, ty = threadIdx.y;
#pragma unroll
  for (int i = 0; i < 4; i++)
    tile[ty + i * 8][tx] = f32_bf16(ip[(size_t)(r0 + ty + i * 8) * cols + c0 + tx]);
  __syncthreads();
#pragma unroll
  for (int i = 0; i < 4; i++)
    op[(size_t)(c0 + ty + i * 8) * rows + r0 + tx] = tile[tx][ty + i * 8];
}

// in [rows, cols] f32 -> out [cols, rows] f32
__global__ __launch_bounds__(256) void k_transpose_f32(const float* __restrict__ in,
                                                       float* __restrict__ out,
                                                       int rows, int cols,
                                                       long long in_bs, long long out_bs) {
  __shared__ float tile[32][33];
  const float* ip = in + (size_t)blockIdx.z * in_bs;
  float* op = out + (size_t)blockIdx.z * out_bs;
  int c0 = blockIdx.x * 32, r0 = blockIdx.y * 32;
  int tx = threadIdx.x, ty = threadIdx.y;
#pragma unroll
  for (int i = 0; i < 4; i++)
    tile[ty + i * 8][tx] = ip[(size_t)(r0 + ty + i * 8) * cols + c0 + tx];
  __syncthreads();
#pragma unroll
  for (int i = 0; i < 4; i++)
    op[(size_t)(c0 + ty + i * 8) * rows + r0 + tx] = tile[tx][ty + i * 8];
}

// ---------- kernel 1: QKV projection (WMMA) ----------
// xb [B,T,C] bf16 ; wtX [N,C] bf16 (transposed weights) ; biases f32.
// which = z%3: 0 -> Q [B,T,K] bf16, 1 -> K [B,T,K] bf16, 2 -> Vt [B,V,T] bf16
__global__ __launch_bounds__(256) void k_qkv(const unsigned short* __restrict__ xb,
                                             const unsigned short* __restrict__ wtq,
                                             const unsigned short* __restrict__ wtk,
                                             const unsigned short* __restrict__ wtv,
                                             const float* __restrict__ bq,
                                             const float* __restrict__ bk,
                                             const float* __restrict__ bv,
                                             unsigned short* __restrict__ Qw,
                                             unsigned short* __restrict__ Kw,
                                             unsigned short* __restrict__ Vt) {
  int z = blockIdx.z, b = z / 3, which = z % 3;
  const unsigned short* wt = (which == 0) ? wtq : (which == 1) ? wtk : wtv;
  const float* bias = (which == 0) ? bq : (which == 1) ? bk : bv;
  int lane = threadIdx.x & 31, wave = threadIdx.x >> 5;
  int m0 = blockIdx.x * 64 + (wave & 1) * 32;
  int n0 = blockIdx.y * 256 + (wave >> 1) * 64;

  const unsigned short* A = xb + (size_t)b * Tdim * Cdim + (size_t)m0 * Cdim;
  const unsigned short* Bp = wt + (size_t)n0 * Cdim;

  v8f acc[2][4];
#pragma unroll
  for (int i = 0; i < 2; i++)
#pragma unroll
    for (int j = 0; j < 4; j++)
#pragma unroll
      for (int e = 0; e < 8; e++) acc[i][j][e] = 0.f;

  gemm_bf16<2, 4>(A, Cdim, Bp, Cdim, Cdim, lane, acc);

#pragma unroll
  for (int i = 0; i < 2; i++)
#pragma unroll
    for (int j = 0; j < 4; j++) {
      int mt = m0 + i * 16 + ((lane >> 4) << 3);
      int nt = n0 + j * 16 + (lane & 15);
      float bi = bias[nt];
#pragma unroll
      for (int r = 0; r < 8; r++) {
        unsigned short h = f32_bf16(acc[i][j][r] + bi);
        if (which == 2)
          Vt[(size_t)b * VD * Tdim + (size_t)nt * Tdim + (mt + r)] = h;
        else if (which == 1)
          Kw[(size_t)b * Tdim * KD + (size_t)(mt + r) * KD + nt] = h;
        else
          Qw[(size_t)b * Tdim * KD + (size_t)(mt + r) * KD + nt] = h;
      }
    }
}

// ---------- kernel 2: scores = scale * Q K^T, causal mask fused ----------
__global__ __launch_bounds__(256) void k_scores(const unsigned short* __restrict__ Qw,
                                                const unsigned short* __restrict__ Kw,
                                                float* __restrict__ w) {
  int b = blockIdx.z;
  int lane = threadIdx.x & 31, wave = threadIdx.x >> 5;
  int m0 = blockIdx.x * 64 + (wave & 1) * 32;   // query t
  int n0 = blockIdx.y * 256 + (wave >> 1) * 64; // key   s
  float* wp = w + (size_t)b * Tdim * Tdim;
  const float scale = 0.04419417382415922f; // 1/sqrt(512)

  if (n0 > m0 + 31) { // wave tile fully masked: no GEMM, fill -inf
#pragma unroll
    for (int i = 0; i < 2; i++)
#pragma unroll
      for (int j = 0; j < 4; j++) {
        int mt = m0 + i * 16 + ((lane >> 4) << 3);
        int nt = n0 + j * 16 + (lane & 15);
#pragma unroll
        for (int r = 0; r < 8; r++)
          wp[(size_t)(mt + r) * Tdim + nt] = -__builtin_inff();
      }
    return;
  }

  const unsigned short* A = Qw + (size_t)b * Tdim * KD + (size_t)m0 * KD;
  const unsigned short* Bp = Kw + (size_t)b * Tdim * KD + (size_t)n0 * KD;

  v8f acc[2][4];
#pragma unroll
  for (int i = 0; i < 2; i++)
#pragma unroll
    for (int j = 0; j < 4; j++)
#pragma unroll
      for (int e = 0; e < 8; e++) acc[i][j][e] = 0.f;

  gemm_bf16<2, 4>(A, KD, Bp, KD, KD, lane, acc);

#pragma unroll
  for (int i = 0; i < 2; i++)
#pragma unroll
    for (int j = 0; j < 4; j++) {
      int mt = m0 + i * 16 + ((lane >> 4) << 3);
      int nt = n0 + j * 16 + (lane & 15);
#pragma unroll
      for (int r = 0; r < 8; r++) {
        int m = mt + r;
        float val = acc[i][j][r] * scale;
        wp[(size_t)m * Tdim + nt] = (nt > m) ? -__builtin_inff() : val;
      }
    }
}

// ---------- kernel 3: softmax over the QUERY axis (per key-column) ----------
__global__ __launch_bounds__(256) void k_softmax_col(float* __restrict__ w) {
  int b = blockIdx.y;
  int s = blockIdx.x * blockDim.x + threadIdx.x;
  float* wp = w + (size_t)b * Tdim * Tdim;
  float m = -__builtin_inff(), sum = 0.f;
  for (int t = 0; t < Tdim; t++) {
    float x = wp[(size_t)t * Tdim + s];
    if (x == -__builtin_inff()) continue;
    if (x > m) { sum = sum * __expf(m - x) + 1.f; m = x; }
    else        sum += __expf(x - m);
  }
  float inv = 1.f / sum;
  for (int t = 0; t < Tdim; t++) {
    float x = wp[(size_t)t * Tdim + s];
    float e = (x == -__builtin_inff()) ? 0.f : __expf(x - m) * inv;
    wp[(size_t)t * Tdim + s] = e;
  }
}

// ---------- kernel 4: out_tmp[t,v] = sum_s weights[t,s] * Vt[v,s] ----------
__global__ __launch_bounds__(256) void k_out(const float* __restrict__ w,
                                             const unsigned short* __restrict__ Vt,
                                             float* __restrict__ outtmp) {
  int b = blockIdx.z;
  int lane = threadIdx.x & 31, wave = threadIdx.x >> 5;
  int m0 = blockIdx.x * 64 + (wave & 1) * 32;   // t
  int n0 = blockIdx.y * 256 + (wave >> 1) * 64; // v

  const float* A = w + (size_t)b * Tdim * Tdim + (size_t)m0 * Tdim;
  const unsigned short* Bp = Vt + (size_t)b * VD * Tdim + (size_t)n0 * Tdim;

  v8f acc[2][4];
#pragma unroll
  for (int i = 0; i < 2; i++)
#pragma unroll
    for (int j = 0; j < 4; j++)
#pragma unroll
      for (int e = 0; e < 8; e++) acc[i][j][e] = 0.f;

  int Keff = m0 + 32; // weights[t,s] == 0 for s > t -> clip the K loop
  gemm_f32A<2, 4>(A, Tdim, Bp, Tdim, Keff, lane, acc);

#pragma unroll
  for (int i = 0; i < 2; i++)
#pragma unroll
    for (int j = 0; j < 4; j++) {
      int mt = m0 + i * 16 + ((lane >> 4) << 3);
      int nt = n0 + j * 16 + (lane & 15);
#pragma unroll
      for (int r = 0; r < 8; r++)
        outtmp[(size_t)b * Tdim * VD + (size_t)(mt + r) * VD + nt] = acc[i][j][r];
    }
}

// ---------- host launcher ----------
extern "C" void kernel_launch(void* const* d_in, const int* in_sizes, int n_in,
                              void* d_out, int out_size, void* d_ws, size_t ws_size,
                              hipStream_t stream) {
  const float* x  = (const float*)d_in[0];
  const float* Wq = (const float*)d_in[1];
  const float* bq = (const float*)d_in[2];
  const float* Wk = (const float*)d_in[3];
  const float* bk = (const float*)d_in[4];
  const float* Wv = (const float*)d_in[5];
  const float* bv = (const float*)d_in[6];
  // d_in[7] = mask (known analytically: s > t masked)

  float* outp = (float*)d_out;                            // [B, V, T]
  float* wp   = outp + (size_t)Bsz * VD * Tdim;           // [B, T, T]

  const size_t nBTC = (size_t)Bsz * Tdim * Cdim;          // 8,388,608
  const size_t nW   = (size_t)Cdim * KD;                  // 262,144
  unsigned short* xb  = (unsigned short*)d_ws;            // [B, T, C] bf16
  unsigned short* wtq = xb  + nBTC;                       // [K, C] bf16
  unsigned short* wtk = wtq + nW;
  unsigned short* wtv = wtk + nW;
  unsigned short* Qw  = wtv + nW;                         // [B, T, K] bf16
  unsigned short* Kw  = Qw  + nBTC;                       // [B, T, K] bf16
  unsigned short* Vt  = Kw  + nBTC;                       // [B, V, T] bf16
  float* outtmp = (float*)(Vt + nBTC);                    // [B, T, V] f32

  dim3 tb(32, 8);
  // x [B][C,T] -> xb [B][T,C] (bf16)
  k_transpose_cvt<<<dim3(Tdim / 32, Cdim / 32, Bsz), tb, 0, stream>>>(
      x, xb, Cdim, Tdim, (long long)Cdim * Tdim, (long long)Tdim * Cdim);
  // W [C,N] -> Wt [N,C] (bf16)
  k_transpose_cvt<<<dim3(KD / 32, Cdim / 32, 1), tb, 0, stream>>>(
      Wq, wtq, Cdim, KD, 0, 0);
  k_transpose_cvt<<<dim3(KD / 32, Cdim / 32, 1), tb, 0, stream>>>(
      Wk, wtk, Cdim, KD, 0, 0);
  k_transpose_cvt<<<dim3(VD / 32, Cdim / 32, 1), tb, 0, stream>>>(
      Wv, wtv, Cdim, VD, 0, 0);

  // QKV projection: block tile 64x256, 8 waves of 32x64
  k_qkv<<<dim3(Tdim / 64, KD / 256, Bsz * 3), dim3(256), 0, stream>>>(
      xb, wtq, wtk, wtv, bq, bk, bv, Qw, Kw, Vt);

  // scores + mask + scale
  k_scores<<<dim3(Tdim / 64, Tdim / 256, Bsz), dim3(256), 0, stream>>>(Qw, Kw, wp);

  // column softmax (axis = query)
  k_softmax_col<<<dim3(Tdim / 256, Bsz), dim3(256), 0, stream>>>(wp);

  // out_tmp = weights @ V  (A fragments converted f32->bf16 in-register)
  k_out<<<dim3(Tdim / 64, VD / 256, Bsz), dim3(256), 0, stream>>>(wp, Vt, outtmp);

  // out_tmp [B][T,V] -> out [B][V,T]
  k_transpose_f32<<<dim3(VD / 32, Tdim / 32, Bsz), tb, 0, stream>>>(
      outtmp, outp, Tdim, VD, (long long)Tdim * VD, (long long)VD * Tdim);
}
